// PathCausalSelfAttention_45174466019393
// MI455X (gfx1250) — matrix operand order repaired and checked by reading, compiled, not run on hardware
//
#include <hip/hip_runtime.h>
#include <hip/hip_bf16.h>
#include <math.h>

#define DEV static __device__ __forceinline__

typedef __attribute__((ext_vector_type(16))) __bf16 bf16x16;
typedef __attribute__((ext_vector_type(8)))  __bf16 bf16x8;
typedef __attribute__((ext_vector_type(8)))  float  f32x8;
typedef __attribute__((ext_vector_type(4)))  int    i32x4;

constexpr int BATCH  = 4;
constexpr int SEQ    = 1024;
constexpr int DMODEL = 1024;
constexpr int NHEAD  = 16;
constexpr int HDIM   = 64;
constexpr int GM     = BATCH * SEQ;           // 4096 rows
// (1/sqrt(hd)) / ln(L)  == 0.125 / ln(1024)
constexpr float ALPHA = 0.018033688f;
constexpr float NEG_BIG = -1.0e30f;

#if defined(__gfx1250__) && __has_builtin(__builtin_amdgcn_global_load_async_to_lds_b128) && __has_builtin(__builtin_amdgcn_s_wait_asynccnt)
#define USE_ASYNC 1
#else
#define USE_ASYNC 0
#endif

DEV bf16x16 join8(bf16x8 lo, bf16x8 hi) {
  return __builtin_shufflevector(lo, hi, 0,1,2,3,4,5,6,7,8,9,10,11,12,13,14,15);
}

DEV f32x8 wmma_bf16(bf16x16 a, bf16x16 b, f32x8 c) {
  // D = A(16x32) * B(32x16) + C, f32 accumulate
  return __builtin_amdgcn_wmma_f32_16x16x32_bf16(false, a, false, b, (short)0, c, false, false);
}

#if USE_ASYNC
typedef __attribute__((address_space(1))) i32x4 gvec_t;
typedef __attribute__((address_space(3))) i32x4 lvec_t;
DEV void async_cp16(const __bf16* g, __bf16* l) {
  // copy 16 bytes global -> LDS via the async DMA path (ASYNCcnt-tracked)
  __builtin_amdgcn_global_load_async_to_lds_b128((gvec_t*)g, (lvec_t*)l, 0, 0);
}
#endif

// ---------------------------------------------------------------------------
// Tiled WMMA GEMM:  C[m,n] = sum_k A[m,k] * Bt[n,k]   (A fp32 MxK, Bt fp32 NxK)
// Block tile 128(M) x 64(N) x 32(K); 8 waves, each wave owns a 32x32 subtile.
// Software-pipelined: global loads for tile kt+1 issue during compute of kt.
// EPI 0: scatter to head-major bf16 Q/K/V   EPI 1: fp32 row-major out
// ---------------------------------------------------------------------------
template <int EPI>
__launch_bounds__(256)
__global__ void gemm_bf16_kernel(const float* __restrict__ A,
                                 const float* __restrict__ Bt,
                                 __bf16* __restrict__ oQ,
                                 __bf16* __restrict__ oK,
                                 __bf16* __restrict__ oV,
                                 float*  __restrict__ oF,
                                 int N, int K) {
  __shared__ __align__(16) __bf16 sA[128][40];  // [m][k], pad 32->40
  __shared__ __align__(16) __bf16 sB[32][72];   // [k][n], pad 64->72

  const int t    = threadIdx.x;
  const int wave = t >> 5;
  const int lane = t & 31;
  const int half = lane >> 4;
  const int nl   = lane & 15;
  const int wm   = wave & 3;     // 4 wave-rows  * 32
  const int wn   = wave >> 2;    // 2 wave-cols  * 32
  const int m0   = blockIdx.y * 128;
  const int n0   = blockIdx.x * 64;

  const int aRow = t >> 1;            // 128 rows, 2 threads/row
  const int aSeg = (t & 1) * 16;      // 16 floats each
  const int bRow = t >> 2;            // 64 rows of Bt
  const int bSeg = (t & 3) * 8;       // 8 floats each

  f32x8 acc[2][2] = {};
  float4 aReg[4], bReg[2];

  auto ldA = [&](int kt) {
    const float* src = A + (size_t)(m0 + aRow) * K + kt + aSeg;
    #pragma unroll
    for (int j = 0; j < 4; ++j) aReg[j] = *(const float4*)(src + 4 * j);
  };
  auto ldB = [&](int kt) {
    const float* src = Bt + (size_t)(n0 + bRow) * K + kt + bSeg;
    bReg[0] = *(const float4*)(src);
    bReg[1] = *(const float4*)(src + 4);
  };

  ldA(0);
  ldB(0);

  for (int kt = 0; kt < K; kt += 32) {
    // --- commit prefetched tile to LDS (fp32 -> bf16)
    #pragma unroll
    for (int j = 0; j < 4; ++j) {
      sA[aRow][aSeg + 4 * j + 0] = (__bf16)aReg[j].x;
      sA[aRow][aSeg + 4 * j + 1] = (__bf16)aReg[j].y;
      sA[aRow][aSeg + 4 * j + 2] = (__bf16)aReg[j].z;
      sA[aRow][aSeg + 4 * j + 3] = (__bf16)aReg[j].w;
    }
    #pragma unroll
    for (int j = 0; j < 2; ++j) {
      sB[bSeg + 4 * j + 0][bRow] = (__bf16)bReg[j].x;
      sB[bSeg + 4 * j + 1][bRow] = (__bf16)bReg[j].y;
      sB[bSeg + 4 * j + 2][bRow] = (__bf16)bReg[j].z;
      sB[bSeg + 4 * j + 3][bRow] = (__bf16)bReg[j].w;
    }
    __syncthreads();

    // --- prefetch next tile while WMMAs run
    if (kt + 32 < K) {
      ldA(kt + 32);
      ldB(kt + 32);
    }

    // --- fragment loads per CDNA5 WMMA VGPR layout
    bf16x16 aF[2], bF[2];
    #pragma unroll
    for (int fm = 0; fm < 2; ++fm) {
      const __bf16* p = &sA[wm * 32 + fm * 16 + nl][half * 8];
      aF[fm] = join8(*(const bf16x8*)p, *(const bf16x8*)(p + 16));
    }
    #pragma unroll
    for (int fn = 0; fn < 2; ++fn) {
      #pragma unroll
      for (int s = 0; s < 16; ++s)
        bF[fn][s] = sB[half * 8 + s + (s & 8)][wn * 32 + fn * 16 + nl];
    }
    #pragma unroll
    for (int fm = 0; fm < 2; ++fm)
      #pragma unroll
      for (int fn = 0; fn < 2; ++fn)
        acc[fm][fn] = wmma_bf16(aF[fm], bF[fn], acc[fm][fn]);
    __syncthreads();
  }

  // --- epilogue: C element (m,n) lives at lane=(m/8)*16 + n%16, vgpr=m%8
  #pragma unroll
  for (int fm = 0; fm < 2; ++fm) {
    #pragma unroll
    for (int fn = 0; fn < 2; ++fn) {
      #pragma unroll
      for (int r = 0; r < 8; ++r) {
        int m = m0 + wm * 32 + fm * 16 + r + 8 * half;
        int n = n0 + wn * 32 + fn * 16 + nl;
        float v = acc[fm][fn][r];
        if constexpr (EPI == 0) {
          // qkv row m = b*SEQ + l ; col n in [0,3D): part -> q/k/v, head-major
          int part = n >> 10;
          int c    = n & 1023;
          int h    = c >> 6;
          int d    = c & 63;
          int b    = m >> 10;
          int l    = m & 1023;
          size_t idx = ((size_t)(b * NHEAD + h) * SEQ + l) * HDIM + d;
          __bf16* dst = (part == 0) ? oQ : (part == 1) ? oK : oV;
          dst[idx] = (__bf16)v;
        } else {
          oF[(size_t)m * N + n] = v;
        }
      }
    }
  }
}

// ---------------------------------------------------------------------------
// q_g / k_g fp32 [B,L,D] -> bf16 head-major [B,H,L,hd]
// ---------------------------------------------------------------------------
__launch_bounds__(256)
__global__ void gates_to_heads_kernel(const float* __restrict__ qg,
                                      const float* __restrict__ kg,
                                      __bf16* __restrict__ QG,
                                      __bf16* __restrict__ KG) {
  int idx = blockIdx.x * 256 + threadIdx.x;   // over B*L*D = 4M
  int d = idx & 63;
  int h = (idx >> 6) & 15;
  int l = (idx >> 10) & 1023;
  int b = idx >> 20;
  size_t dst = ((size_t)(b * NHEAD + h) * SEQ + l) * HDIM + d;
  QG[dst] = (__bf16)qg[idx];
  KG[dst] = (__bf16)kg[idx];
}

// ---------------------------------------------------------------------------
// Flash attention, dual-path scores, causal, per-(b,h) 64-query tiles.
// 4 waves; wave w owns query rows [64*qTile + 16w, +16). Key blocks of 32.
// K/KG prefetched in registers (transposed VALU store); V (and Q/QG) staged
// via GLOBAL_LOAD_ASYNC_TO_LDS with a double-buffered sV when available.
// ---------------------------------------------------------------------------
__launch_bounds__(128)
__global__ void attn_kernel(const __bf16* __restrict__ Q,
                            const __bf16* __restrict__ K,
                            const __bf16* __restrict__ V,
                            const __bf16* __restrict__ QG,
                            const __bf16* __restrict__ KG,
                            float* __restrict__ Y) {
  const int bh    = blockIdx.y;            // 0..63
  const int b     = bh >> 4;
  const int h     = bh & 15;
  const int qTile = blockIdx.x;            // 0..15
  const int t     = threadIdx.x;
  const int w     = t >> 5;
  const int lane  = t & 31;
  const int half  = lane >> 4;
  const int nl    = lane & 15;

  __shared__ __align__(16) __bf16 sQ  [64][72];    // [q][d]
  __shared__ __align__(16) __bf16 sQG [64][72];
  __shared__ __align__(16) __bf16 sKt [64][40];    // [d][key]  (transposed)
  __shared__ __align__(16) __bf16 sKGt[64][40];
  __shared__ __align__(16) __bf16 sV  [2][32][72]; // [buf][key][d]
  __shared__ __align__(16) __bf16 sP  [4][16][40]; // per-wave P staging

  const size_t headBase = (size_t)bh * SEQ * HDIM;
  const int qBase = qTile * 64;

  // --- Q/QG tile (64x64 bf16 each), loaded once
#if USE_ASYNC
  {
    const int row = t >> 3;
    const int off = (t & 7) * 8;
    #pragma unroll
    for (int c = 0; c < 4; ++c) {
      async_cp16(Q  + headBase + (size_t)(qBase + c * 16 + row) * HDIM + off,
                 &sQ[c * 16 + row][off]);
      async_cp16(QG + headBase + (size_t)(qBase + c * 16 + row) * HDIM + off,
                 &sQG[c * 16 + row][off]);
    }
  }
#else
  {
    const int q  = t >> 1;
    const int c0 = (t & 1) * 32;
    const __bf16* gq  = Q  + headBase + (size_t)(qBase + q) * HDIM + c0;
    const __bf16* gqg = QG + headBase + (size_t)(qBase + q) * HDIM + c0;
    #pragma unroll
    for (int j = 0; j < 32; j += 8) {
      *(bf16x8*)&sQ [q][c0 + j] = *(const bf16x8*)(gq  + j);
      *(bf16x8*)&sQG[q][c0 + j] = *(const bf16x8*)(gqg + j);
    }
  }
#endif

  f32x8 accO[4] = {};
  float mrow[8], lrow[8];
  #pragma unroll
  for (int r = 0; r < 8; ++r) { mrow[r] = NEG_BIG; lrow[r] = 0.0f; }

  const int rowG0 = qBase + w * 16 + 8 * half;   // + r = global query row
  const int nKb = 2 * (qTile + 1);

  // --- K/KG register prefetch + V staging helpers
  const int key = t >> 2;             // 32 keys, 4 threads/key
  const int c0  = (t & 3) * 16;       // 16 d-elements each
  bf16x8 kr[2], kgr[2];

  auto loadK = [&](int kb) {
    const __bf16* gk = K  + headBase + (size_t)(kb * 32 + key) * HDIM + c0;
    const __bf16* gg = KG + headBase + (size_t)(kb * 32 + key) * HDIM + c0;
    kr[0]  = *(const bf16x8*)(gk);
    kr[1]  = *(const bf16x8*)(gk + 8);
    kgr[0] = *(const bf16x8*)(gg);
    kgr[1] = *(const bf16x8*)(gg + 8);
  };
#if USE_ASYNC
  auto loadV = [&](int kb, int buf) {
    const int vkey = t >> 3;          // 16 keys per call
    const int voff = (t & 7) * 8;
    #pragma unroll
    for (int c = 0; c < 2; ++c)
      async_cp16(V + headBase + (size_t)(kb * 32 + c * 16 + vkey) * HDIM + voff,
                 &sV[buf][c * 16 + vkey][voff]);
  };
#else
  auto loadV = [&](int kb, int buf) {
    const __bf16* gv = V + headBase + (size_t)(kb * 32 + key) * HDIM + c0;
    *(bf16x8*)&sV[buf][key][c0 + 0] = *(const bf16x8*)(gv + 0);
    *(bf16x8*)&sV[buf][key][c0 + 8] = *(const bf16x8*)(gv + 8);
  };
#endif

  loadK(0);
#if USE_ASYNC
  loadV(0, 0);
#endif

  for (int kb = 0; kb < nKb; ++kb) {
    const int cur = kb & 1;
#if USE_ASYNC
    __builtin_amdgcn_s_wait_asynccnt(0);   // V(kb) (and first-iter Q/QG) landed
#endif
    __syncthreads();   // previous iteration's reads of sKt/sV done; async data visible

    // --- commit prefetched K/KG registers transposed into LDS
    #pragma unroll
    for (int j = 0; j < 8; ++j) {
      sKt [c0 + j    ][key] = kr[0][j];
      sKt [c0 + 8 + j][key] = kr[1][j];
      sKGt[c0 + j    ][key] = kgr[0][j];
      sKGt[c0 + 8 + j][key] = kgr[1][j];
    }
#if !USE_ASYNC
    loadV(kb, cur);
#endif
    __syncthreads();

    // --- prefetch next key block while WMMAs run
    if (kb + 1 < nKb) {
      loadK(kb + 1);
#if USE_ASYNC
      loadV(kb + 1, cur ^ 1);
#endif
    }

    // --- S = (Q K^T + QG KG^T), two 16-key fragments
    f32x8 s[2];
    #pragma unroll
    for (int fn = 0; fn < 2; ++fn) {
      f32x8 c = {};
      #pragma unroll
      for (int ks = 0; ks < 2; ++ks) {      // hd=64 -> two K=32 steps
        const __bf16* pa = &sQ[w * 16 + nl][ks * 32 + half * 8];
        bf16x16 a = join8(*(const bf16x8*)pa, *(const bf16x8*)(pa + 16));
        bf16x16 bf;
        #pragma unroll
        for (int s = 0; s < 16; ++s)
          bf[s] = sKt[ks * 32 + half * 8 + s + (s & 8)][fn * 16 + nl];
        c = wmma_bf16(a, bf, c);
        const __bf16* pag = &sQG[w * 16 + nl][ks * 32 + half * 8];
        bf16x16 ag = join8(*(const bf16x8*)pag, *(const bf16x8*)(pag + 16));
        bf16x16 bg;
        #pragma unroll
        for (int s = 0; s < 16; ++s)
          bg[s] = sKGt[ks * 32 + half * 8 + s + (s & 8)][fn * 16 + nl];
        c = wmma_bf16(ag, bg, c);
      }
      s[fn] = c;
    }

    // --- scale + causal mask (element (m,n): lane=(m/8)*16+n, vgpr=m%8)
    #pragma unroll
    for (int fn = 0; fn < 2; ++fn) {
      int keyG = kb * 32 + fn * 16 + nl;
      #pragma unroll
      for (int r = 0; r < 8; ++r) {
        int rowG = rowG0 + r;
        float v = s[fn][r] * ALPHA;
        s[fn][r] = (keyG > rowG) ? NEG_BIG : v;
      }
    }

    // --- online softmax (row reductions across the 16-lane half)
    float mnew[8], scl[8];
    #pragma unroll
    for (int r = 0; r < 8; ++r) {
      float v = fmaxf(s[0][r], s[1][r]);
      #pragma unroll
      for (int off = 1; off < 16; off <<= 1)
        v = fmaxf(v, __shfl_xor(v, off, 32));
      mnew[r] = fmaxf(mrow[r], v);
      scl[r]  = __expf(mrow[r] - mnew[r]);
      mrow[r] = mnew[r];
    }
    #pragma unroll
    for (int fn = 0; fn < 2; ++fn)
      #pragma unroll
      for (int r = 0; r < 8; ++r)
        s[fn][r] = __expf(s[fn][r] - mnew[r]);
    #pragma unroll
    for (int r = 0; r < 8; ++r) {
      float v = s[0][r] + s[1][r];
      #pragma unroll
      for (int off = 1; off < 16; off <<= 1)
        v += __shfl_xor(v, off, 32);
      lrow[r] = lrow[r] * scl[r] + v;
    }
    #pragma unroll
    for (int dg = 0; dg < 4; ++dg)
      #pragma unroll
      for (int r = 0; r < 8; ++r)
        accO[dg][r] *= scl[r];

    // --- P -> LDS (re-fragment C layout into A layout)
    #pragma unroll
    for (int fn = 0; fn < 2; ++fn)
      #pragma unroll
      for (int r = 0; r < 8; ++r)
        sP[w][r + 8 * half][fn * 16 + nl] = (__bf16)s[fn][r];
    __syncthreads();

    // --- O += P V  (K=32 contraction over this key block)
    {
      const __bf16* pp = &sP[w][nl][half * 8];
      bf16x16 aP = join8(*(const bf16x8*)pp, *(const bf16x8*)(pp + 16));
      #pragma unroll
      for (int dg = 0; dg < 4; ++dg) {
        bf16x16 bV;
        #pragma unroll
        for (int s = 0; s < 16; ++s)
          bV[s] = sV[cur][half * 8 + s + (s & 8)][dg * 16 + nl];
        accO[dg] = wmma_bf16(aP, bV, accO[dg]);
      }
    }
  }

  // --- finalize: O /= l, merge heads into Y[b, l, h*64 + d] (fp32)
  #pragma unroll
  for (int dg = 0; dg < 4; ++dg) {
    #pragma unroll
    for (int r = 0; r < 8; ++r) {
      int rowG = rowG0 + r;
      int col  = h * HDIM + dg * 16 + nl;
      Y[((size_t)b * SEQ + rowG) * DMODEL + col] = accO[dg][r] / lrow[r];
    }
  }
}

// ---------------------------------------------------------------------------
extern "C" void kernel_launch(void* const* d_in, const int* in_sizes, int n_in,
                              void* d_out, int out_size, void* d_ws, size_t ws_size,
                              hipStream_t stream) {
  const float* x    = (const float*)d_in[0];
  const float* qg   = (const float*)d_in[1];
  const float* kg   = (const float*)d_in[2];
  const float* Wqkv = (const float*)d_in[3];
  const float* Wout = (const float*)d_in[4];
  float* out = (float*)d_out;

  const size_t NE = (size_t)BATCH * NHEAD * SEQ * HDIM;   // 4M elems
  __bf16* Qb  = (__bf16*)d_ws;
  __bf16* Kb  = Qb + NE;
  __bf16* Vb  = Qb + 2 * NE;
  __bf16* QGb = Qb + 3 * NE;
  __bf16* KGb = Qb + 4 * NE;
  float*  Yf  = (float*)((char*)d_ws + 5 * NE * sizeof(__bf16));

  // gate repack (independent of gemm1)
  gates_to_heads_kernel<<<dim3((BATCH * SEQ * DMODEL) / 256), dim3(256), 0, stream>>>(
      qg, kg, QGb, KGb);

  // qkv = x @ Wqkv^T  -> head-major bf16 Q/K/V
  gemm_bf16_kernel<0><<<dim3((3 * DMODEL) / 64, GM / 128), dim3(256), 0, stream>>>(
      x, Wqkv, Qb, Kb, Vb, nullptr, 3 * DMODEL, DMODEL);

  // flash attention
  attn_kernel<<<dim3(SEQ / 64, BATCH * NHEAD), dim3(128), 0, stream>>>(
      Qb, Kb, Vb, QGb, KGb, Yf);

  // out = Y @ Wout^T -> fp32 d_out
  gemm_bf16_kernel<1><<<dim3(DMODEL / 64, GM / 128), dim3(256), 0, stream>>>(
      Yf, Wout, nullptr, nullptr, nullptr, out, DMODEL, DMODEL);
}
